// GraphEmbedding_34892314313071
// MI455X (gfx1250) — matrix operand order repaired and checked by reading, not compiled, not run on hardware
//
#include <hip/hip_runtime.h>
#include <hip/hip_bf16.h>

// ---------------------------------------------------------------------------
// Types for CDNA5 WMMA (wave32): 16x16x32 bf16 -> f32 accumulate
// ---------------------------------------------------------------------------
typedef __attribute__((ext_vector_type(16))) __bf16 v16bf;
typedef __attribute__((ext_vector_type(8)))  __bf16 v8bf;
typedef __attribute__((ext_vector_type(8)))  float  v8f;

#define D_FEAT 384
#define N_REL  8

// ---------------------------------------------------------------------------
// Weight prep: Wt[m][n][k] = (bf16) W_m[k][n]   (transpose so B-fragment loads
// are contiguous in K per output column). m: 0..7 = W_rel[r], 8 = W_mp, 9 = W_self
// ---------------------------------------------------------------------------
__global__ void prep_weights_kernel(const float* __restrict__ Wrel,
                                    const float* __restrict__ Wmp,
                                    const float* __restrict__ Wself,
                                    __bf16* __restrict__ Wt)
{
    int idx = blockIdx.x * blockDim.x + threadIdx.x;       // over 10*384*384
    if (idx >= 10 * D_FEAT * D_FEAT) return;
    int m   = idx / (D_FEAT * D_FEAT);
    int rem = idx % (D_FEAT * D_FEAT);
    int n   = rem / D_FEAT;
    int k   = rem % D_FEAT;
    const float* src = (m < 8) ? (Wrel + (size_t)m * D_FEAT * D_FEAT)
                               : (m == 8 ? Wmp : Wself);
    Wt[idx] = (__bf16)src[(size_t)k * D_FEAT + n];
}

// ---------------------------------------------------------------------------
// Node prep: rnorm[i] = rsqrt(sum x[i]^2), hb = bf16(x). One wave per node.
// ---------------------------------------------------------------------------
__global__ __launch_bounds__(256)
void node_prep_kernel(const float* __restrict__ x,
                      __bf16* __restrict__ hb,
                      float* __restrict__ rnorm, int N)
{
    int node = blockIdx.x * (blockDim.x >> 5) + (threadIdx.x >> 5);
    int lane = threadIdx.x & 31;
    if (node >= N) return;
    const float* xr = x + (size_t)node * D_FEAT;
    __bf16* hr = hb + (size_t)node * D_FEAT;
    float s = 0.f;
#pragma unroll
    for (int j = 0; j < D_FEAT / 32; ++j) {
        float v = xr[lane + 32 * j];
        s += v * v;
        hr[lane + 32 * j] = (__bf16)v;
    }
#pragma unroll
    for (int off = 16; off > 0; off >>= 1) s += __shfl_xor(s, off, 32);
    if (lane == 0) rnorm[node] = rsqrtf(s);
}

// ---------------------------------------------------------------------------
// Per-edge cosine similarity. One wave per edge.
// ---------------------------------------------------------------------------
__global__ __launch_bounds__(256)
void edge_cos_kernel(const float* __restrict__ x,
                     const int* __restrict__ ei,    // [2,E] row0=src row1=dst
                     const float* __restrict__ rnorm,
                     float* __restrict__ cos_e, int E)
{
    int e    = blockIdx.x * (blockDim.x >> 5) + (threadIdx.x >> 5);
    int lane = threadIdx.x & 31;
    if (e >= E) return;
    int s = ei[e], d = ei[E + e];
    const float* xs = x + (size_t)s * D_FEAT;
    const float* xd = x + (size_t)d * D_FEAT;
    float acc = 0.f;
#pragma unroll
    for (int j = 0; j < D_FEAT / 32; ++j)
        acc += xs[lane + 32 * j] * xd[lane + 32 * j];
#pragma unroll
    for (int off = 16; off > 0; off >>= 1) acc += __shfl_xor(acc, off, 32);
    if (lane == 0) cos_e[e] = acc * rnorm[s] * rnorm[d];
}

// ---------------------------------------------------------------------------
// WMMA GEMM: C[M,384] = A_bf16[M,384] @ W + bias   (W given as Wt = W^T bf16)
// One wave computes a 16x64 tile: one A fragment feeds 4 WMMAs per K step
// (4x less A refetch vs one-tile-per-wave). K fully unrolled: 48 v_wmma/wave.
//   A fragment (16-bit A, 16x32): lane(0..15)=row M, elems 0..7 = K(k0+hs*8..),
//   elems 8..15 = K(k0+16+hs*8..), hs = lane>>4.
//   B fragment: lane(0..15)=col N, elems 0..7 = K(k0+hs*8..),
//   elems 8..15 = K(k0+16+hs*8..), hs = lane>>4 (same as A).
// ---------------------------------------------------------------------------
__global__ __launch_bounds__(256)
void gemm_bf16_kernel(const __bf16* __restrict__ A,
                      const __bf16* __restrict__ Bt,
                      const float*  __restrict__ bias,
                      float* __restrict__ C, int M)
{
    const int lane    = threadIdx.x & 31;
    const int wave    = threadIdx.x >> 5;
    const int nGroups = D_FEAT / 64;                        // 6 groups of 4 tiles
    const int tile    = blockIdx.x * (blockDim.x >> 5) + wave;
    const int mTile   = tile / nGroups;
    const int nGroup  = tile % nGroups;
    if (mTile * 16 >= M) return;

    const int hs  = lane >> 4;                              // lane half select
    const int l15 = lane & 15;
    const __bf16* aRow = A + (size_t)(mTile * 16 + l15) * D_FEAT + hs * 8;
    const __bf16* bRow0 = Bt + (size_t)(nGroup * 64 + l15) * D_FEAT + hs * 8;
    // 4 adjacent N tiles: columns nGroup*64 + t*16 + l15, rows stride 16*384
    const size_t bTileStride = (size_t)16 * D_FEAT;

    v8f acc0 = {}, acc1 = {}, acc2 = {}, acc3 = {};
#pragma unroll
    for (int k0 = 0; k0 < D_FEAT; k0 += 32) {
        v8bf a0 = *(const v8bf*)(aRow + k0);
        v8bf a1 = *(const v8bf*)(aRow + k0 + 16);
        v16bf a;
#pragma unroll
        for (int i = 0; i < 8; ++i) { a[i] = a0[i]; a[8 + i] = a1[i]; }

#pragma unroll
        for (int t = 0; t < 4; ++t) {
            const __bf16* bp = bRow0 + (size_t)t * bTileStride + k0;
            v8bf b0 = *(const v8bf*)(bp);
            v8bf b1 = *(const v8bf*)(bp + 16);
            v16bf b;
#pragma unroll
            for (int i = 0; i < 8; ++i) { b[i] = b0[i]; b[8 + i] = b1[i]; }
            v8f& acc = (t == 0) ? acc0 : (t == 1) ? acc1 : (t == 2) ? acc2 : acc3;
            acc = __builtin_amdgcn_wmma_f32_16x16x32_bf16(
                /*neg_a=*/false, a, /*neg_b=*/false, b,
                /*c_mod=*/(short)0, acc, /*reuse_a=*/false, /*reuse_b=*/false);
        }
    }

#pragma unroll
    for (int t = 0; t < 4; ++t) {
        const v8f& acc = (t == 0) ? acc0 : (t == 1) ? acc1 : (t == 2) ? acc2 : acc3;
        const int col  = nGroup * 64 + t * 16 + l15;
        const float bv = bias[col];
        float* cBase = C + (size_t)(mTile * 16 + hs * 8) * D_FEAT + col;
#pragma unroll
        for (int i = 0; i < 8; ++i)
            cBase[(size_t)i * D_FEAT] = acc[i] + bv;        // row = mTile*16+hs*8+i
    }
}

// ---------------------------------------------------------------------------
// Relation scatter: for edges with etype==r:
//   acc[dst] += (cos_e / norm_c[r]) * lin[src]   (wave per edge, f32 atomics)
// ---------------------------------------------------------------------------
__global__ __launch_bounds__(256)
void scatter_rel_kernel(const float* __restrict__ lin,
                        float* __restrict__ acc,
                        const int* __restrict__ ei,
                        const int* __restrict__ etype,
                        const float* __restrict__ cos_e,
                        const float* __restrict__ norm_c,
                        int r, int E)
{
    int e    = blockIdx.x * (blockDim.x >> 5) + (threadIdx.x >> 5);
    int lane = threadIdx.x & 31;
    if (e >= E) return;
    if (etype[e] != r) return;
    int s = ei[e], d = ei[E + e];
    float w = cos_e[e] / norm_c[r];
    const float* lp = lin + (size_t)s * D_FEAT;
    float*       op = acc + (size_t)d * D_FEAT;
#pragma unroll
    for (int j = 0; j < D_FEAT / 32; ++j)
        atomicAdd(&op[lane + 32 * j], w * lp[lane + 32 * j]);
}

// ---------------------------------------------------------------------------
// MP scatter: acc[dst] += lin[src] for all edges.
// ---------------------------------------------------------------------------
__global__ __launch_bounds__(256)
void scatter_mp_kernel(const float* __restrict__ lin,
                       float* __restrict__ acc,
                       const int* __restrict__ ei, int E)
{
    int e    = blockIdx.x * (blockDim.x >> 5) + (threadIdx.x >> 5);
    int lane = threadIdx.x & 31;
    if (e >= E) return;
    int s = ei[e], d = ei[E + e];
    const float* lp = lin + (size_t)s * D_FEAT;
    float*       op = acc + (size_t)d * D_FEAT;
#pragma unroll
    for (int j = 0; j < D_FEAT / 32; ++j)
        atomicAdd(&op[lane + 32 * j], lp[lane + 32 * j]);
}

// ---------------------------------------------------------------------------
// relu in place on acc + bf16 cast into hb
// ---------------------------------------------------------------------------
__global__ void relu_cast_kernel(float* __restrict__ acc,
                                 __bf16* __restrict__ hb, size_t n)
{
    size_t i = (size_t)blockIdx.x * blockDim.x + threadIdx.x;
    if (i >= n) return;
    float v = acc[i];
    v = v > 0.f ? v : 0.f;
    acc[i] = v;
    hb[i]  = (__bf16)v;
}

__global__ void fill_zero_kernel(float* __restrict__ p, size_t n)
{
    size_t i = (size_t)blockIdx.x * blockDim.x + threadIdx.x;
    if (i < n) p[i] = 0.f;
}

// ---------------------------------------------------------------------------
// Mean pool over nodes-per-graph: out[g][c] = mean_n h[g*nper+n][c]
// ---------------------------------------------------------------------------
__global__ void pool_mean_kernel(const float* __restrict__ h,
                                 float* __restrict__ out, int nper)
{
    int g = blockIdx.x;
    int c = threadIdx.x;                                    // blockDim = 384
    const float* base = h + ((size_t)g * nper) * D_FEAT + c;
    float s = 0.f;
    for (int n = 0; n < nper; ++n) s += base[(size_t)n * D_FEAT];
    out[(size_t)g * D_FEAT + c] = s * (1.0f / (float)nper);
}

// ---------------------------------------------------------------------------
// Host launcher
// ---------------------------------------------------------------------------
extern "C" void kernel_launch(void* const* d_in, const int* in_sizes, int n_in,
                              void* d_out, int out_size, void* d_ws, size_t ws_size,
                              hipStream_t stream)
{
    const float* x      = (const float*)d_in[0];   // [N,384]
    const int*   ei     = (const int*)  d_in[1];   // [2,E]
    const int*   etype  = (const int*)  d_in[2];   // [E]
    const float* Wrel   = (const float*)d_in[4];   // [8,384,384]
    const float* brel   = (const float*)d_in[5];   // [8,384]
    const float* normc  = (const float*)d_in[6];   // [8]
    const float* Wmp    = (const float*)d_in[7];   // [384,384]
    const float* bmp    = (const float*)d_in[8];   // [384]
    const float* Wself  = (const float*)d_in[9];   // [384,384]
    const float* bself  = (const float*)d_in[10];  // [384]

    const int N    = in_sizes[0] / D_FEAT;         // 32768
    const int E    = in_sizes[2];                  // 524288
    const int B    = out_size / D_FEAT;            // 128
    const int nper = N / B;                        // 256

    // ---- carve workspace (256B aligned slabs) ----
    char* p = (char*)d_ws;
    auto carve = [&](size_t bytes) {
        void* q = (void*)p;
        p += (bytes + 255) & ~(size_t)255;
        return q;
    };
    __bf16* Wt    = (__bf16*)carve((size_t)10 * D_FEAT * D_FEAT * sizeof(__bf16));
    __bf16* hb    = (__bf16*)carve((size_t)N * D_FEAT * sizeof(__bf16));
    float*  accb  = (float*) carve((size_t)N * D_FEAT * sizeof(float));
    float*  lin   = (float*) carve((size_t)N * D_FEAT * sizeof(float));
    float*  rnorm = (float*) carve((size_t)N * sizeof(float));
    float*  cose  = (float*) carve((size_t)E * sizeof(float));
    (void)ws_size; (void)n_in;

    const size_t nd = (size_t)N * D_FEAT;

    // ---- prep ----
    {
        int total = 10 * D_FEAT * D_FEAT;
        prep_weights_kernel<<<(total + 255) / 256, 256, 0, stream>>>(Wrel, Wmp, Wself, Wt);
    }
    node_prep_kernel<<<(N + 7) / 8, 256, 0, stream>>>(x, hb, rnorm, N);
    edge_cos_kernel<<<(E + 7) / 8, 256, 0, stream>>>(x, ei, rnorm, cose, E);

    const int gemmBlocks  = ((N / 16) * (D_FEAT / 64) + 7) / 8;   // 8 waves/block
    const int edgeBlocks  = (E + 7) / 8;
    const int elemBlocks  = (int)((nd + 255) / 256);

    for (int layer = 0; layer < 2; ++layer) {
        // ---- relation-aware message passing ----
        fill_zero_kernel<<<elemBlocks, 256, 0, stream>>>(accb, nd);
        for (int r = 0; r < N_REL; ++r) {
            gemm_bf16_kernel<<<gemmBlocks, 256, 0, stream>>>(
                hb, Wt + (size_t)r * D_FEAT * D_FEAT, brel + (size_t)r * D_FEAT, lin, N);
            scatter_rel_kernel<<<edgeBlocks, 256, 0, stream>>>(
                lin, accb, ei, etype, cose, normc, r, E);
        }
        relu_cast_kernel<<<elemBlocks, 256, 0, stream>>>(accb, hb, nd);

        // ---- MP layer: acc = h@W_self + b_self + segsum(h@W_mp + b_mp)[src->dst]
        gemm_bf16_kernel<<<gemmBlocks, 256, 0, stream>>>(
            hb, Wt + (size_t)8 * D_FEAT * D_FEAT, bmp, lin, N);
        gemm_bf16_kernel<<<gemmBlocks, 256, 0, stream>>>(
            hb, Wt + (size_t)9 * D_FEAT * D_FEAT, bself, accb, N);
        scatter_mp_kernel<<<edgeBlocks, 256, 0, stream>>>(lin, accb, ei, E);
        relu_cast_kernel<<<elemBlocks, 256, 0, stream>>>(accb, hb, nd);
    }

    // ---- mean pooling ----
    pool_mean_kernel<<<B, D_FEAT, 0, stream>>>(accb, (float*)d_out, nper);
}
